// CoevolutionAnalyzer_38414187495428
// MI455X (gfx1250) — compile-verified, matrix operand order checked by reading, not run on hardware
//
#include <hip/hip_runtime.h>

typedef __attribute__((ext_vector_type(16))) _Float16 v16h;
typedef __attribute__((ext_vector_type(8)))  float    v8f;

namespace {
constexpr int D    = 128;                 // feature dim
constexpr int H    = 64;                  // hidden dim (D/2)
constexpr int SEQ  = 512;
constexpr int BATCH = 2;
constexpr int NT   = SEQ / 16;            // 32 16-wide tiles per axis
constexpr int NTRI = NT * (NT + 1) / 2;   // 528 upper-triangular tile pairs
}

// ---------------------------------------------------------------------------
// Kernel 1: g = dom*evo ; a = g @ W1[:D] + b1 ; c = g @ W1[D:]
// b1 is folded into a so the hot fragment build needs only ONE add per elem.
// ---------------------------------------------------------------------------
__global__ __launch_bounds__(D) void coevo_prep(
    const float* __restrict__ dom, const float* __restrict__ evo,
    const float* __restrict__ W1, const float* __restrict__ b1,
    float* __restrict__ aBuf, float* __restrict__ cBuf) {
  __shared__ float sg[D];
  const int bn = blockIdx.x;            // b*SEQ + n
  const int e  = threadIdx.x;           // output feature
  sg[e] = dom[bn * D + e] * evo[bn * D + e];
  __syncthreads();
  float va = b1[e], vc = 0.f;           // bias folded into a
#pragma unroll 8
  for (int d = 0; d < D; ++d) {
    const float gv = sg[d];
    va = fmaf(gv, W1[d * D + e], va);          // W1[:D]  row d
    vc = fmaf(gv, W1[(D + d) * D + e], vc);    // W1[D:]  row d
  }
  aBuf[bn * D + e] = va;
  cBuf[bn * D + e] = vc;
}

// ---------------------------------------------------------------------------
// Kernel 2: repack W2 [128 x 64] fp32 -> f16 WMMA B-fragments.
// B layout mirrors the documented 16-bit A layout (CDNA5 ISA §7.12.2):
//   lane holds column n = (lane&15), K half chosen by lane>=16 (kbase = 8),
//   element e -> K = kbase + (e&7) + (e>=8 ? 16 : 0) within the 32-K chunk.
// ---------------------------------------------------------------------------
__global__ __launch_bounds__(512) void coevo_pack_w2(
    const float* __restrict__ W2, _Float16* __restrict__ w2f) {
  const int tid   = threadIdx.x;        // tid = (kc*4+nc)*32 + lane
  const int lane  = tid & 31;
  const int nc    = (tid >> 5) & 3;
  const int kc    = tid >> 7;
  const int kbase = (lane >> 4) * 8;
  const int n     = nc * 16 + (lane & 15);
#pragma unroll
  for (int e = 0; e < 16; ++e) {
    const int k = kc * 32 + kbase + (e & 7) + ((e >> 3) << 4);
    w2f[tid * 16 + e] = (_Float16)W2[k * H + n];
  }
}

// ---------------------------------------------------------------------------
// Kernel 3: fused pairwise MLP over one 16x16 (i,j) tile per workgroup.
// 16 wave32s; wave w owns row i = ti*16 + w, its 16 j's are the WMMA M dim.
// ---------------------------------------------------------------------------
__global__ __launch_bounds__(512) void coevo_main(
    const float* __restrict__ aBuf, const float* __restrict__ cBuf,
    const _Float16* __restrict__ w2f, const float* __restrict__ b2,
    const float* __restrict__ W3, const float* __restrict__ b3,
    float* __restrict__ out) {
  __shared__ float sA[16][D];           // a rows (b1 pre-folded)
  __shared__ float sC[16][D];
  __shared__ float sB2[H];
  __shared__ float sW3[H];
  __shared__ __align__(32) _Float16 sW2[4 * 4 * 32 * 16];  // 8192 halfs = 16 KB

  const int b = blockIdx.y;
  // decode upper-triangular tile index: tiles (ti, tj) with tj >= ti
  int rem = blockIdx.x, ti = 0;
  while (rem >= NT - ti) { rem -= NT - ti; ++ti; }
  const int tj = ti + rem;

  const int tid = threadIdx.x;
  // --- cooperative staging ---
#pragma unroll
  for (int r = 0; r < 4; ++r) {
    const int idx = tid + r * 512;             // 0..2047
    const int row = idx >> 7, k = idx & (D - 1);
    sA[row][k] = aBuf[((size_t)b * SEQ + ti * 16 + row) * D + k];
    sC[row][k] = cBuf[((size_t)b * SEQ + tj * 16 + row) * D + k];
  }
#pragma unroll
  for (int r = 0; r < 8; ++r)                  // 8192 halfs as 4096 dwords
    ((unsigned int*)sW2)[tid + r * 512] = ((const unsigned int*)w2f)[tid + r * 512];
  if (tid < H)            sB2[tid] = b2[tid];
  else if (tid < 2 * H)   sW3[tid - H] = W3[tid - H];
  __syncthreads();

  const int lane  = tid & 31;
  const int wave  = tid >> 5;                  // 0..15 -> i row in tile
  const int mrow  = lane & 15;                 // A-fragment row (j index m)
  const int kbase = (lane >> 4) * 8;

  v8f acc[4] = {};                             // 4 N-tiles of 16 outputs
#pragma unroll
  for (int kc = 0; kc < 4; ++kc) {
    // Build A fragment: h[m,k] = relu((a_i[k]+b1[k]) + c_j[k]) -> f16
    v16h afrag;
#pragma unroll
    for (int e = 0; e < 16; ++e) {
      const int k = kc * 32 + kbase + (e & 7) + ((e >> 3) << 4);
      const float v = sA[wave][k] + sC[mrow][k];
      afrag[e] = (_Float16)fmaxf(v, 0.f);
    }
#pragma unroll
    for (int nc = 0; nc < 4; ++nc) {
      const v16h bfrag = *(const v16h*)&sW2[((kc * 4 + nc) * 32 + lane) * 16];
      acc[nc] = __builtin_amdgcn_wmma_f32_16x16x32_f16(
          false, afrag, false, bfrag, (short)0, acc[nc], false, false);
    }
  }

  // --- epilogue: relu(h2 + b2) . w3, sigmoid, symmetric scatter ---
  const int nlane = lane & 15;
  const float b3v = b3[0];
  const int i = ti * 16 + wave;
  float* o = out + (size_t)b * SEQ * SEQ;
#pragma unroll
  for (int r = 0; r < 8; ++r) {
    float p = 0.f;
#pragma unroll
    for (int nc = 0; nc < 4; ++nc) {
      const int n = nc * 16 + nlane;
      p += fmaxf(acc[nc][r] + sB2[n], 0.f) * sW3[n];
    }
    // reduce the 16 N-lanes within each lane-half (M differs between halves)
#pragma unroll
    for (int off = 1; off < 16; off <<= 1) p += __shfl_xor(p, off, 32);
    if (nlane == 0) {
      const int m = r + ((lane >> 4) << 3);    // C/D layout: lanes>=16 hold M=r+8
      const int j = tj * 16 + m;
      const float s = 1.f / (1.f + __expf(-(p + b3v)));
      if (i < j) {
        o[i * SEQ + j] = s;
        o[j * SEQ + i] = s;
      } else if (i == j) {
        o[i * SEQ + j] = 0.f;                  // diagonal stays zero
      }
      // i > j pairs (only in diagonal tiles) are discarded; their locations
      // are written by the mirrored (j,i) pair.
    }
  }
}

// ---------------------------------------------------------------------------
extern "C" void kernel_launch(void* const* d_in, const int* in_sizes, int n_in,
                              void* d_out, int out_size, void* d_ws, size_t ws_size,
                              hipStream_t stream) {
  (void)in_sizes; (void)n_in; (void)out_size; (void)ws_size;
  const float* dom = (const float*)d_in[0];
  const float* evo = (const float*)d_in[1];
  const float* W1  = (const float*)d_in[2];
  const float* b1  = (const float*)d_in[3];
  const float* W2  = (const float*)d_in[4];
  const float* b2  = (const float*)d_in[5];
  const float* W3  = (const float*)d_in[6];
  const float* b3  = (const float*)d_in[7];
  float* out = (float*)d_out;

  // workspace: a+b1 [B*SEQ*D] f32 | c [B*SEQ*D] f32 | W2 fragments f16 (16 KB)
  float* aBuf = (float*)d_ws;
  float* cBuf = aBuf + (size_t)BATCH * SEQ * D;
  _Float16* w2f = (_Float16*)(cBuf + (size_t)BATCH * SEQ * D);

  coevo_prep<<<BATCH * SEQ, D, 0, stream>>>(dom, evo, W1, b1, aBuf, cBuf);
  coevo_pack_w2<<<1, 512, 0, stream>>>(W2, w2f);
  coevo_main<<<dim3(NTRI, BATCH), 512, 0, stream>>>(aBuf, cBuf, w2f, b2, W3,
                                                    b3, out);
}
